// MultiHeadAttention_78692390797830
// MI455X (gfx1250) — matrix operand order repaired and checked by reading, not compiled, not run on hardware
//
#include <hip/hip_runtime.h>
#include <hip/hip_bf16.h>

typedef __bf16 bf16;
typedef __attribute__((ext_vector_type(16))) __bf16 v16bf;
typedef __attribute__((ext_vector_type(8)))  __bf16 v8bf;
typedef __attribute__((ext_vector_type(8)))  float   v8f;

#define DIM   2048
#define NH    16
#define HD    128
#define BATCH 2
#define SEQ   2048
#define MROWS (BATCH * SEQ)

#if defined(__gfx1250__)
#define USE_ASYNC_COPY 1
#else
#define USE_ASYNC_COPY 0
#endif

// ---------------------------------------------------------------------------
// WMMA fragment helpers (wave32, layouts per CDNA5 ISA 7.12.2)
// ---------------------------------------------------------------------------

// Combine two 8-element chunks: elements j<8 at p, j>=8 at p+16 elements.
__device__ __forceinline__ v16bf frag_split16(const bf16* p) {
    v8bf lo = *(const v8bf*)p;
    v8bf hi = *(const v8bf*)(p + 16);
    v16bf a;
#pragma unroll
    for (int j = 0; j < 8; ++j) { a[j] = lo[j]; a[j + 8] = hi[j]; }
    return a;
}
// Combine one contiguous 16-element chunk at p.
__device__ __forceinline__ v16bf frag_contig16(const bf16* p) {
    v8bf lo = *(const v8bf*)p;
    v8bf hi = *(const v8bf*)(p + 8);
    v16bf b;
#pragma unroll
    for (int j = 0; j < 8; ++j) { b[j] = lo[j]; b[j + 8] = hi[j]; }
    return b;
}

// A-matrix 16x32 bf16, row-major source: K = k0 + 8*half + {0..7, 16..23}
__device__ __forceinline__ v16bf load_a_frag(const bf16* __restrict__ base, int ld,
                                             int row0, int k0, int lane) {
    const int r = lane & 15, half = lane >> 4;
    return frag_split16(base + (size_t)(row0 + r) * ld + k0 + half * 8);
}

// B-matrix 32x16 bf16 held as B^T row-major (Bt[n][k]): K = k0 + 8*half + {0..7, 16..23}
__device__ __forceinline__ v16bf load_bt_frag(const bf16* __restrict__ base, int ld,
                                              int n0, int k0, int lane) {
    const int n = lane & 15, half = lane >> 4;
    return frag_split16(base + (size_t)(n0 + n) * ld + k0 + half * 8);
}

__device__ __forceinline__ v8f wmma_bf16(v16bf a, v16bf b, v8f c) {
    return __builtin_amdgcn_wmma_f32_16x16x32_bf16(false, a, false, b, (short)0, c,
                                                   false, false);
}

__device__ __forceinline__ float hmax16(float v) {
#pragma unroll
    for (int off = 1; off < 16; off <<= 1) v = fmaxf(v, __shfl_xor(v, off, 32));
    return v;
}
__device__ __forceinline__ float hsum16(float v) {
#pragma unroll
    for (int off = 1; off < 16; off <<= 1) v += __shfl_xor(v, off, 32);
    return v;
}

// ---------------------------------------------------------------------------
// Kernel 0: fp32 -> bf16 pack
// ---------------------------------------------------------------------------
__global__ void pack_bf16(const float* __restrict__ src, bf16* __restrict__ dst, int n) {
    int i = blockIdx.x * blockDim.x + threadIdx.x;
    if (i < n) dst[i] = (bf16)src[i];
}

// ---------------------------------------------------------------------------
// Kernel 1/3: Y = X @ W^T  (M=MROWS, N=DIM, K=DIM).
// Block = 4 waves sharing one 64-row M strip, double-buffer staged in LDS via
// global_load_async_to_lds_b128 (ASYNCcnt) overlapping compute; each wave owns
// a 64-wide N strip -> 4x4 tiles of 16x16 per wave, 16 WMMAs per K=32 step.
// mode 0: Q -> RoPE, bf16 [b][h][s][d];  mode 1: K -> same
// mode 2: V -> bf16 [b][h][d][s] (transposed);  mode 3: f32 [m][n]
// ---------------------------------------------------------------------------
#define LDSROW 40   // padded row stride (elements) to stagger LDS banks

__global__ void __launch_bounds__(128, 1)
qkv_gemm(const bf16* __restrict__ X, const bf16* __restrict__ W,
         void* __restrict__ outp,
         const float* __restrict__ fcos, const float* __restrict__ fsin,
         int mode) {
    const int lane = threadIdx.x;
    const int m0g  = blockIdx.x * 64;
    const int n0w  = (blockIdx.y * 4 + threadIdx.y) * 64;

#if USE_ASYNC_COPY
    __shared__ __align__(16) bf16 As[2][64][LDSROW];
    const int tid     = threadIdx.y * 32 + threadIdx.x;   // 0..127
    const int rowc    = tid >> 1;                          // row this thread copies
    const int halfsel = tid & 1;                           // which 32B of the row
    const bf16* gXrow = X + (size_t)(m0g + rowc) * DIM + 16 * halfsel;
#endif

    v8f acc[4][4];
#pragma unroll
    for (int i = 0; i < 4; ++i)
#pragma unroll
        for (int j = 0; j < 4; ++j) acc[i][j] = (v8f){};

    const int r = lane & 15, half = lane >> 4;

#if USE_ASYNC_COPY
    // prologue: stage first A tile into buffer 0
    {
        unsigned lds0 = (unsigned)(uintptr_t)&As[0][rowc][16 * halfsel];
        unsigned long long g0 = (unsigned long long)(uintptr_t)gXrow;
        asm volatile("global_load_async_to_lds_b128 %0, %1, off"
                     :: "v"(lds0), "v"(g0) : "memory");
        asm volatile("global_load_async_to_lds_b128 %0, %1, off offset:16"
                     :: "v"(lds0 + 16), "v"(g0) : "memory");
    }
#endif

#pragma unroll 1
    for (int kt = 0; kt < DIM / 32; ++kt) {
        const int k = kt * 32;
#if USE_ASYNC_COPY
        const int cur = kt & 1;
        // Drain my copies into As[cur], then barrier: after this, every wave's
        // copies into As[cur] are complete, and all reads of As[cur^1] from the
        // previous iteration are behind us.
        asm volatile("s_wait_asynccnt 0x0" ::: "memory");
        __syncthreads();
        if (kt + 1 < DIM / 32) {   // kick off next tile into the other buffer
            unsigned lds0 = (unsigned)(uintptr_t)&As[cur ^ 1][rowc][16 * halfsel];
            unsigned long long g0 =
                (unsigned long long)(uintptr_t)(gXrow + (size_t)(k + 32));
            asm volatile("global_load_async_to_lds_b128 %0, %1, off"
                         :: "v"(lds0), "v"(g0) : "memory");
            asm volatile("global_load_async_to_lds_b128 %0, %1, off offset:16"
                         :: "v"(lds0 + 16), "v"(g0) : "memory");
        }

        v16bf a[4];
#pragma unroll
        for (int i = 0; i < 4; ++i)
            a[i] = frag_split16(&As[cur][16 * i + r][half * 8]);
#else
        v16bf a[4];
#pragma unroll
        for (int i = 0; i < 4; ++i)
            a[i] = load_a_frag(X, DIM, m0g + 16 * i, k, lane);
#endif

#pragma unroll
        for (int j = 0; j < 4; ++j) {
            v16bf b = load_bt_frag(W, DIM, n0w + 16 * j, k, lane);
#pragma unroll
            for (int i = 0; i < 4; ++i)
                acc[i][j] = wmma_bf16(a[i], b, acc[i][j]);
        }
    }

    // ---- epilogue ----
    const int col = lane & 15;
#pragma unroll
    for (int i = 0; i < 4; ++i) {
        const int m0 = m0g + 16 * i;
#pragma unroll
        for (int j = 0; j < 4; ++j) {
            const int n = n0w + 16 * j + col;
            if (mode == 3) {
                float* O = (float*)outp;
#pragma unroll
                for (int rr = 0; rr < 8; ++rr) {
                    int m = m0 + rr + 8 * half;
                    O[(size_t)m * DIM + n] = acc[i][j][rr];
                }
            } else {
                bf16* O = (bf16*)outp;
                const int h = n >> 7, d = n & (HD - 1);
#pragma unroll
                for (int rr = 0; rr < 8; ++rr) {
                    int m = m0 + rr + 8 * half;
                    int bb = m >> 11, s = m & (SEQ - 1);
                    float v = acc[i][j][rr];
                    if (mode <= 1) {   // RoPE: pair partner sits in adjacent lane
                        float pr = __shfl_xor(v, 1, 32);
                        float cs = fcos[(size_t)s * (HD / 2) + (d >> 1)];
                        float sn = fsin[(size_t)s * (HD / 2) + (d >> 1)];
                        v = (d & 1) ? (pr * sn + v * cs) : (v * cs - pr * sn);
                    }
                    if (mode == 2)
                        O[((size_t)(bb * NH + h) * HD + d) * SEQ + s] = (bf16)v;  // V^T
                    else
                        O[((size_t)(bb * NH + h) * SEQ + s) * HD + d] = (bf16)v;  // Q/K
                }
            }
        }
    }
}

// ---------------------------------------------------------------------------
// Kernel 2: flash attention, one wave per (b,h,q-tile of 16 rows).
// 32-key chunks: 8 WMMAs Q@K^T + 8 WMMAs P@V per chunk.
// ---------------------------------------------------------------------------
__global__ void __launch_bounds__(32, 1)
flash_attn(const bf16* __restrict__ Q, const bf16* __restrict__ K,
           const bf16* __restrict__ Vt, const float* __restrict__ mask,
           bf16* __restrict__ Ob) {
    __shared__ __align__(16) bf16 pbuf[16][32];

    const int lane  = threadIdx.x;
    const int qtile = blockIdx.x;            // 0..SEQ/16-1
    const int bh    = blockIdx.y;            // 0..BATCH*NH-1
    const int bidx = bh >> 4, h = bh & 15;
    const int q0 = qtile * 16;
    const int col = lane & 15, half = lane >> 4;

    const bf16* Qh = Q  + (size_t)bh * SEQ * HD;
    const bf16* Kh = K  + (size_t)bh * SEQ * HD;
    const bf16* Vh = Vt + (size_t)bh * HD * SEQ;

    v16bf qa[4];
#pragma unroll
    for (int i = 0; i < 4; ++i) qa[i] = load_a_frag(Qh, HD, q0, i * 32, lane);

    v8f o[8];
#pragma unroll
    for (int i = 0; i < 8; ++i) o[i] = (v8f){};
    float mrow[8], lrow[8];
#pragma unroll
    for (int r = 0; r < 8; ++r) { mrow[r] = -1e30f; lrow[r] = 0.0f; }

    const float scale = 0.08838834764831845f;  // 1/sqrt(HD)

#pragma unroll 1
    for (int kb = 0; kb < SEQ; kb += 32) {
        v8f s0 = {}, s1 = {};
#pragma unroll
        for (int i = 0; i < 4; ++i)
            s0 = wmma_bf16(qa[i], load_bt_frag(Kh, HD, kb,      i * 32, lane), s0);
#pragma unroll
        for (int i = 0; i < 4; ++i)
            s1 = wmma_bf16(qa[i], load_bt_frag(Kh, HD, kb + 16, i * 32, lane), s1);

        float alpha[8];
#pragma unroll
        for (int r = 0; r < 8; ++r) {
            int sq = q0 + r + 8 * half;
            float x0 = s0[r] * scale + mask[(size_t)sq * SEQ + kb + col];
            float x1 = s1[r] * scale + mask[(size_t)sq * SEQ + kb + 16 + col];
            float cm   = hmax16(fmaxf(x0, x1));
            float mnew = fmaxf(mrow[r], cm);
            float al   = __expf(mrow[r] - mnew);
            mrow[r] = mnew;
            float p0 = __expf(x0 - mnew);
            float p1 = __expf(x1 - mnew);
            lrow[r] = lrow[r] * al + hsum16(p0 + p1);
            alpha[r] = al;
            pbuf[r + 8 * half][col]      = (bf16)p0;
            pbuf[r + 8 * half][col + 16] = (bf16)p1;
        }
        asm volatile("s_wait_dscnt 0x0" ::: "memory");

        v16bf pa = frag_split16(&pbuf[col][half * 8]);

#pragma unroll
        for (int nt = 0; nt < 8; ++nt) {
#pragma unroll
            for (int r = 0; r < 8; ++r) o[nt][r] *= alpha[r];
            v16bf vb = load_bt_frag(Vh, SEQ, nt * 16, kb, lane);
            o[nt] = wmma_bf16(pa, vb, o[nt]);
        }
    }

#pragma unroll
    for (int r = 0; r < 8; ++r) {
        int sq = q0 + r + 8 * half;
        float inv = 1.0f / lrow[r];
        size_t rowoff = ((size_t)bidx * SEQ + sq) * DIM + (size_t)h * HD;
#pragma unroll
        for (int nt = 0; nt < 8; ++nt)
            Ob[rowoff + nt * 16 + col] = (bf16)(o[nt][r] * inv);
    }
}

// ---------------------------------------------------------------------------
// Host orchestration
// ---------------------------------------------------------------------------
extern "C" void kernel_launch(void* const* d_in, const int* in_sizes, int n_in,
                              void* d_out, int out_size, void* d_ws, size_t ws_size,
                              hipStream_t stream) {
    const float* x    = (const float*)d_in[0];
    const float* fcos = (const float*)d_in[1];
    const float* fsin = (const float*)d_in[2];
    const float* mask = (const float*)d_in[3];
    const float* wq   = (const float*)d_in[4];
    const float* wk   = (const float*)d_in[5];
    const float* wv   = (const float*)d_in[6];
    const float* wo   = (const float*)d_in[7];
    (void)in_sizes; (void)n_in; (void)out_size; (void)ws_size;

    const size_t SZX = (size_t)MROWS * DIM;
    const size_t SZW = (size_t)DIM * DIM;

    char* ws = (char*)d_ws;
    bf16* Xb  = (bf16*)ws;  ws += SZX * sizeof(bf16);
    bf16* Wqb = (bf16*)ws;  ws += SZW * sizeof(bf16);
    bf16* Wkb = (bf16*)ws;  ws += SZW * sizeof(bf16);
    bf16* Wvb = (bf16*)ws;  ws += SZW * sizeof(bf16);
    bf16* Wob = (bf16*)ws;  ws += SZW * sizeof(bf16);
    bf16* Qb  = (bf16*)ws;  ws += SZX * sizeof(bf16);
    bf16* Kb  = (bf16*)ws;  ws += SZX * sizeof(bf16);
    bf16* Vtb = (bf16*)ws;  ws += SZX * sizeof(bf16);
    bf16* Ab  = (bf16*)ws;  ws += SZX * sizeof(bf16);

    pack_bf16<<<(int)((SZX + 255) / 256), 256, 0, stream>>>(x,  Xb,  (int)SZX);
    pack_bf16<<<(int)((SZW + 255) / 256), 256, 0, stream>>>(wq, Wqb, (int)SZW);
    pack_bf16<<<(int)((SZW + 255) / 256), 256, 0, stream>>>(wk, Wkb, (int)SZW);
    pack_bf16<<<(int)((SZW + 255) / 256), 256, 0, stream>>>(wv, Wvb, (int)SZW);
    pack_bf16<<<(int)((SZW + 255) / 256), 256, 0, stream>>>(wo, Wob, (int)SZW);

    // 64-row M strips x (4 waves * 64-col N strips)
    dim3 gg(MROWS / 64, DIM / 256);
    dim3 gb(32, 4);
    qkv_gemm<<<gg, gb, 0, stream>>>(Xb, Wqb, Qb,  fcos, fsin, 0);
    qkv_gemm<<<gg, gb, 0, stream>>>(Xb, Wkb, Kb,  fcos, fsin, 1);
    qkv_gemm<<<gg, gb, 0, stream>>>(Xb, Wvb, Vtb, fcos, fsin, 2);

    flash_attn<<<dim3(SEQ / 16, BATCH * NH), 32, 0, stream>>>(Qb, Kb, Vtb, mask, Ab);

    qkv_gemm<<<gg, gb, 0, stream>>>(Ab, Wob, d_out, fcos, fsin, 3);
}
